// CrossRPEAttentionMulti_43044162240871
// MI455X (gfx1250) — matrix operand order repaired and checked by reading, not compiled, not run on hardware
//
#include <hip/hip_runtime.h>
#include <hip/hip_bf16.h>

// ---------------------------------------------------------------------------
// CrossRPE attention for gfx1250 (MI455X), wave32 + WMMA f32_16x16x32_f16.
// Pipeline: convert->f16, QKV GEMM (WMMA + async-LDS DMA double buffering),
// qt=q@rpe, fused flash attention (QK^T + bucket bias + online softmax + PV,
// all WMMA), proj GEMM (same async WMMA core).
// ---------------------------------------------------------------------------

typedef __attribute__((ext_vector_type(16))) _Float16 v16h;
typedef __attribute__((ext_vector_type(8)))  _Float16 v8h;
typedef __attribute__((ext_vector_type(8)))  float    v8f;

#define B_    8
#define NH    12
#define HD    64
#define DIM_  768
#define SEQ   1025
#define NR    1056          // padded sequence rows (33*32)
#define MR    8320          // padded B_*SEQ rows (65*128)
#define MTOT  8200          // B_*SEQ
#define NBUK  8
#define LDT   40            // LDS tile stride (f16 elems), 80B rows (16B aligned)

__device__ __forceinline__ v8f wmma_ff(v16h a, v16h b, v8f c) {
  // D = A(16x32 f16) * B(32x16 f16) + C(16x16 f32)
  return __builtin_amdgcn_wmma_f32_16x16x32_f16(false, a, false, b, (short)0, c,
                                                false, false);
}

// A-matrix 16x32 f16 fragment. Lanes 0-15: row=lane, K={0..7,16..23};
// lanes 16-31: row=lane-16, K={8..15,24..31}. `base` = row-major tile, ld elems.
__device__ __forceinline__ v16h afrag(const _Float16* base, int ld, int lane) {
  const int row = lane & 15;
  const int ko  = (lane >> 4) << 3;           // 0 or 8
  const _Float16* p = base + row * ld + ko;
  union { v16h v; v8h h[2]; } u;
  u.h[0] = *(const v8h*)(p);
  u.h[1] = *(const v8h*)(p + 16);
  return u.v;
}

// B-matrix 32x16 f16 fragment from a row-major B^T tile (n-major, ld elems).
// Lanes 0-15: col n=lane, K={0..7,16..23}; lanes 16-31: col n=lane-16, K={8..15,24..31}.
__device__ __forceinline__ v16h bfrag(const _Float16* base, int ld, int lane) {
  const int n  = lane & 15;
  const int ko = (lane >> 4) << 3;            // 0 or 8
  const _Float16* p = base + n * ld + ko;
  union { v16h v; v8h h[2]; } u;
  u.h[0] = *(const v8h*)(p);
  u.h[1] = *(const v8h*)(p + 16);
  return u.v;
}

// ---- gfx1250 async cache->LDS DMA (ASYNCcnt-tracked, no VGPR round trip) ---
__device__ __forceinline__ unsigned lds_off(const void* p) {
  // generic shared ptr = SHARED_BASE[63:32] | lds_offset[31:0]
  return (unsigned)(unsigned long long)p;
}
__device__ __forceinline__ void async_b128(unsigned lds, const void* gaddr) {
  asm volatile("global_load_async_to_lds_b128 %0, %1, off"
               :: "v"(lds), "v"(gaddr) : "memory");
}
#define ASYNC_WAIT_3() asm volatile("s_wait_asynccnt 0x3" ::: "memory")
#define ASYNC_WAIT_0() asm volatile("s_wait_asynccnt 0x0" ::: "memory")

// ---------------------------------------------------------------------------
// 1) fp32 -> f16 staging for X, qkv_w, proj_w (pad rows of X zeroed)
// ---------------------------------------------------------------------------
__global__ void k_convert(const float* __restrict__ x,
                          const float* __restrict__ qkv_w,
                          const float* __restrict__ proj_w,
                          _Float16* __restrict__ X16,
                          _Float16* __restrict__ W16,
                          _Float16* __restrict__ PW16) {
  const long T0 = (long)MR * DIM_;
  const long T1 = (long)3 * DIM_ * DIM_;
  const long T2 = (long)DIM_ * DIM_;
  const long T  = T0 + T1 + T2;
  const long stride = (long)gridDim.x * blockDim.x;
  for (long t = (long)blockIdx.x * blockDim.x + threadIdx.x; t < T; t += stride) {
    if (t < T0) {
      long m = t / DIM_;
      X16[t] = (m < MTOT) ? (_Float16)x[t] : (_Float16)0.f;
    } else if (t < T0 + T1) {
      long u = t - T0;
      W16[u] = (_Float16)qkv_w[u];
    } else {
      long u = t - T0 - T1;
      PW16[u] = (_Float16)proj_w[u];
    }
  }
}

// ---------------------------------------------------------------------------
// Shared double-buffered WMMA GEMM core: block tile 128(M) x 64(N), K step 32,
// 8 waves each owning one 16-row sub-tile. Tiles DMA'd to LDS asynchronously.
// ---------------------------------------------------------------------------
struct Smem {
  _Float16 A[2][128 * LDT];
  _Float16 B[2][64 * LDT];
};

__device__ __forceinline__ void issue_tile(const _Float16* gA, const _Float16* gB,
                                           Smem& sm, int buf, int m0, int n0,
                                           int k0, int tid) {
  // A tile 128x32 f16 = 512 x b128 chunks -> 2 per thread
  {
    const int c0 = tid * 2;
    int row = c0 >> 2, seg = c0 & 3;
    async_b128(lds_off(&sm.A[buf][row * LDT + seg * 8]),
               gA + (size_t)(m0 + row) * DIM_ + k0 + seg * 8);
    const int c1 = c0 + 1;
    row = c1 >> 2; seg = c1 & 3;
    async_b128(lds_off(&sm.A[buf][row * LDT + seg * 8]),
               gA + (size_t)(m0 + row) * DIM_ + k0 + seg * 8);
  }
  // B tile 64x32 f16 = 256 x b128 chunks -> 1 per thread
  {
    const int row = tid >> 2, seg = tid & 3;
    async_b128(lds_off(&sm.B[buf][row * LDT + seg * 8]),
               gB + (size_t)(n0 + row) * DIM_ + k0 + seg * 8);
  }
}

__device__ __forceinline__ void gemm_core(const _Float16* __restrict__ gA,
                                          const _Float16* __restrict__ gB,
                                          Smem& sm, int m0, int n0, int tid,
                                          v8f acc[4]) {
  const int lane = tid & 31;
  const int wave = tid >> 5;
  issue_tile(gA, gB, sm, 0, m0, n0, 0, tid);
  int buf = 0;
  for (int k0 = 0; k0 < DIM_; k0 += 32, buf ^= 1) {
    if (k0 + 32 < DIM_) {
      issue_tile(gA, gB, sm, buf ^ 1, m0, n0, k0 + 32, tid);
      ASYNC_WAIT_3();          // in-order async: <=3 outstanding => tile k done
    } else {
      ASYNC_WAIT_0();
    }
    __syncthreads();
    v16h af = afrag(&sm.A[buf][wave * 16 * LDT], LDT, lane);
#pragma unroll
    for (int t = 0; t < 4; ++t) {
      v16h bf = bfrag(&sm.B[buf][t * 16 * LDT], LDT, lane);
      acc[t] = wmma_ff(af, bf, acc[t]);
    }
    __syncthreads();           // all waves done reading before buf is re-DMA'd
  }
}

// ---------------------------------------------------------------------------
// 2) QKV GEMM: [MR x 768] x [2304 x 768]^T. Epilogue scatters into
//    Q16/K16 [b,h,NR,64] (Q scaled by d^-0.5) and V transposed [b,h,64,NR].
// ---------------------------------------------------------------------------
__global__ __launch_bounds__(256) void k_gemm_qkv(
    const _Float16* __restrict__ X16, const _Float16* __restrict__ W16,
    const float* __restrict__ qkv_b,
    _Float16* __restrict__ Q16, _Float16* __restrict__ K16,
    _Float16* __restrict__ VT16) {
  __shared__ Smem sm;
  const int tid  = threadIdx.x;
  const int lane = tid & 31;
  const int wave = tid >> 5;
  const int m0 = blockIdx.y * 128;
  const int n0 = blockIdx.x * 64;

  v8f acc[4] = {};
  gemm_core(X16, W16, sm, m0, n0, tid, acc);

  const int hi = (lane >> 4) & 1;
  const int nl = lane & 15;
#pragma unroll
  for (int t = 0; t < 4; ++t) {
    const int n = n0 + t * 16 + nl;
    const float bias = qkv_b[n];
    const int part = n / DIM_;
    const int c = n - part * DIM_;
    const int hh = c >> 6, dc = c & 63;
#pragma unroll
    for (int r = 0; r < 8; ++r) {
      const int m = m0 + wave * 16 + r + hi * 8;
      if (m >= MTOT) continue;
      const float v = acc[t][r] + bias;
      const int b = m / SEQ, i = m - b * SEQ;
      const size_t bh = (size_t)(b * NH + hh);
      if (part == 0)      Q16[(bh * NR + i) * HD + dc]  = (_Float16)(v * 0.125f);
      else if (part == 1) K16[(bh * NR + i) * HD + dc]  = (_Float16)v;
      else                VT16[(bh * HD + dc) * NR + i] = (_Float16)v;
    }
  }
}

// ---------------------------------------------------------------------------
// 3) qt[b,h,i,u] = q[b,h,i,:] . rpe[h,u,:]   (tiny: N*nb dot-64s)
// ---------------------------------------------------------------------------
__global__ void k_qt(const _Float16* __restrict__ Q16,
                     const float* __restrict__ rpe_rows,
                     const float* __restrict__ rpe_cols,
                     float* __restrict__ QTR, float* __restrict__ QTC, int nb) {
  const long t = (long)blockIdx.x * blockDim.x + threadIdx.x;
  const long T = (long)B_ * NH * SEQ * NBUK;
  if (t >= T) return;
  const int  u  = (int)(t & (NBUK - 1));
  const long r  = t >> 3;
  const int  i  = (int)(r % SEQ);
  const long bh = r / SEQ;
  const int  hh = (int)(bh % NH);
  const _Float16* q = Q16 + (bh * NR + i) * HD;
  const float* wr = rpe_rows + ((size_t)hh * nb + u) * HD;
  const float* wc = rpe_cols + ((size_t)hh * nb + u) * HD;
  float sr = 0.f, sc = 0.f;
#pragma unroll 8
  for (int k = 0; k < HD; ++k) {
    const float qv = (float)q[k];
    sr += qv * wr[k];
    sc += qv * wc[k];
  }
  QTR[(bh * NR + i) * NBUK + u] = sr;
  QTC[(bh * NR + i) * NBUK + u] = sc;
}

// ---------------------------------------------------------------------------
// 4) Fused flash attention: one wave per (b,h, 16-query-row tile).
//    S = Q K^T (WMMA) + bucket-gather bias, online softmax, O += P V (WMMA).
// ---------------------------------------------------------------------------
__global__ __launch_bounds__(32) void k_flash(
    const _Float16* __restrict__ Q16, const _Float16* __restrict__ K16,
    const _Float16* __restrict__ VT16,
    const float* __restrict__ QTR, const float* __restrict__ QTC,
    const int* __restrict__ brow, const int* __restrict__ bcol,
    _Float16* __restrict__ AO16) {
  __shared__ _Float16 Pl[16 * LDT];                // 16x32 P staging
  const int lane = threadIdx.x;
  const int hi = (lane >> 4) & 1;
  const int nl = lane & 15;
  const int bid = blockIdx.x;
  const int mt = bid % 65;                          // 65 row tiles of 16
  const int bh = bid / 65;
  const int b  = bh / NH, hh = bh % NH;
  const int i0 = mt * 16;

  const _Float16* Q  = Q16  + (size_t)bh * NR * HD;
  const _Float16* K  = K16  + (size_t)bh * NR * HD;
  const _Float16* VT = VT16 + (size_t)bh * HD * NR;
  const float* qtr = QTR + (size_t)bh * NR * NBUK;
  const float* qtc = QTC + (size_t)bh * NR * NBUK;

  // Q fragments for the full d=64 reduction (2 x K=32), kept resident.
  const v16h aq0 = afrag(Q + (size_t)i0 * HD,      HD, lane);
  const v16h aq1 = afrag(Q + (size_t)i0 * HD + 32, HD, lane);

  float mrow[8], lrow[8];
#pragma unroll
  for (int r = 0; r < 8; ++r) { mrow[r] = -1e30f; lrow[r] = 0.f; }
  v8f oacc[4] = {};

  for (int j0 = 0; j0 < NR; j0 += 32) {
    v8f s0 = {}, s1 = {};
    {
      v16h bk  = bfrag(K + (size_t)j0 * HD,      HD, lane);
      v16h bk2 = bfrag(K + (size_t)j0 * HD + 32, HD, lane);
      s0 = wmma_ff(aq0, bk,  s0);
      s0 = wmma_ff(aq1, bk2, s0);
      bk  = bfrag(K + (size_t)(j0 + 16) * HD,      HD, lane);
      bk2 = bfrag(K + (size_t)(j0 + 16) * HD + 32, HD, lane);
      s1 = wmma_ff(aq0, bk,  s1);
      s1 = wmma_ff(aq1, bk2, s1);
    }
    // contextual RPE bias gather + sequence mask
#pragma unroll
    for (int r = 0; r < 8; ++r) {
      const int ig  = i0 + r + hi * 8;
      const int jg0 = j0 + nl;
      const int jg1 = j0 + 16 + nl;
      if (ig < SEQ && jg0 < SEQ) {
        const int bu = brow[(size_t)ig * SEQ + jg0];
        const int bv = bcol[(size_t)ig * SEQ + jg0];
        s0[r] += qtr[ig * NBUK + bu] + qtc[ig * NBUK + bv];
      } else s0[r] = -1e30f;
      if (ig < SEQ && jg1 < SEQ) {
        const int bu = brow[(size_t)ig * SEQ + jg1];
        const int bv = bcol[(size_t)ig * SEQ + jg1];
        s1[r] += qtr[ig * NBUK + bu] + qtc[ig * NBUK + bv];
      } else s1[r] = -1e30f;
    }
    // online softmax; rows live across 16-lane halves -> xor-tree reductions
#pragma unroll
    for (int r = 0; r < 8; ++r) {
      float sm = fmaxf(s0[r], s1[r]);
      sm = fmaxf(sm, __shfl_xor(sm, 1, 32));
      sm = fmaxf(sm, __shfl_xor(sm, 2, 32));
      sm = fmaxf(sm, __shfl_xor(sm, 4, 32));
      sm = fmaxf(sm, __shfl_xor(sm, 8, 32));
      const float mnew = fmaxf(mrow[r], sm);
      const float corr = __expf(mrow[r] - mnew);
      const float p0 = __expf(s0[r] - mnew);
      const float p1 = __expf(s1[r] - mnew);
      float ps = p0 + p1;
      ps += __shfl_xor(ps, 1, 32);
      ps += __shfl_xor(ps, 2, 32);
      ps += __shfl_xor(ps, 4, 32);
      ps += __shfl_xor(ps, 8, 32);
      lrow[r] = lrow[r] * corr + ps;
      mrow[r] = mnew;
#pragma unroll
      for (int t = 0; t < 4; ++t) oacc[t][r] *= corr;
      s0[r] = p0; s1[r] = p1;
    }
    // P (16x16 f32 C-layout) -> LDS -> A-fragment layout for PV
#pragma unroll
    for (int r = 0; r < 8; ++r) {
      const int row = r + hi * 8;
      Pl[row * LDT + nl]      = (_Float16)s0[r];
      Pl[row * LDT + 16 + nl] = (_Float16)s1[r];
    }
    __syncthreads();
    const v16h pf = afrag(Pl, LDT, lane);
#pragma unroll
    for (int t = 0; t < 4; ++t) {
      v16h bv = bfrag(VT + (size_t)(t * 16) * NR + j0, NR, lane);
      oacc[t] = wmma_ff(pf, bv, oacc[t]);
    }
    __syncthreads();
  }
  // normalize and write [B,N,h*d] f16 for the projection GEMM
#pragma unroll
  for (int t = 0; t < 4; ++t) {
#pragma unroll
    for (int r = 0; r < 8; ++r) {
      const int ig = i0 + r + hi * 8;
      if (ig >= SEQ) continue;
      const float v = oacc[t][r] / lrow[r];
      const size_t mg = (size_t)b * SEQ + ig;
      AO16[mg * DIM_ + hh * HD + t * 16 + nl] = (_Float16)v;
    }
  }
}

// ---------------------------------------------------------------------------
// 5) Output projection: [MR x 768] x [768 x 768]^T + proj_b -> f32 out
// ---------------------------------------------------------------------------
__global__ __launch_bounds__(256) void k_gemm_proj(
    const _Float16* __restrict__ A16, const _Float16* __restrict__ PW16,
    const float* __restrict__ proj_b, float* __restrict__ out) {
  __shared__ Smem sm;
  const int tid  = threadIdx.x;
  const int lane = tid & 31;
  const int wave = tid >> 5;
  const int m0 = blockIdx.y * 128;
  const int n0 = blockIdx.x * 64;

  v8f acc[4] = {};
  gemm_core(A16, PW16, sm, m0, n0, tid, acc);

  const int hi = (lane >> 4) & 1;
  const int nl = lane & 15;
#pragma unroll
  for (int t = 0; t < 4; ++t) {
    const int n = n0 + t * 16 + nl;
    const float bias = proj_b[n];
#pragma unroll
    for (int r = 0; r < 8; ++r) {
      const int m = m0 + wave * 16 + r + hi * 8;
      if (m >= MTOT) continue;
      out[(size_t)m * DIM_ + n] = acc[t][r] + bias;
    }
  }
}

// ---------------------------------------------------------------------------
extern "C" void kernel_launch(void* const* d_in, const int* in_sizes, int n_in,
                              void* d_out, int out_size, void* d_ws, size_t ws_size,
                              hipStream_t stream) {
  (void)n_in; (void)out_size; (void)ws_size;
  const float* x      = (const float*)d_in[0];
  const float* qkv_w  = (const float*)d_in[1];
  const float* qkv_b  = (const float*)d_in[2];
  const float* proj_w = (const float*)d_in[3];
  const float* proj_b = (const float*)d_in[4];
  const float* rpe_r  = (const float*)d_in[5];
  const float* rpe_c  = (const float*)d_in[6];
  const int*   brow   = (const int*)d_in[7];
  const int*   bcol   = (const int*)d_in[8];
  const int nb = in_sizes[5] / (NH * HD);          // == 8

  char* ws = (char*)d_ws;
  size_t off = 0;
  auto take = [&](size_t bytes) -> void* {
    void* p = ws + off;
    off += (bytes + 255) & ~(size_t)255;
    return p;
  };
  _Float16* Q16  = (_Float16*)take((size_t)B_ * NH * NR * HD * 2);
  _Float16* K16  = (_Float16*)take((size_t)B_ * NH * NR * HD * 2);
  _Float16* VT16 = (_Float16*)take((size_t)B_ * NH * NR * HD * 2);
  _Float16* X16  = (_Float16*)take((size_t)MR * DIM_ * 2);
  _Float16* W16  = (_Float16*)take((size_t)3 * DIM_ * DIM_ * 2);
  _Float16* PW16 = (_Float16*)take((size_t)DIM_ * DIM_ * 2);
  _Float16* AO16 = (_Float16*)take((size_t)MR * DIM_ * 2);
  float*    QTR  = (float*)take((size_t)B_ * NH * NR * NBUK * 4);
  float*    QTC  = (float*)take((size_t)B_ * NH * NR * NBUK * 4);

  k_convert<<<2048, 256, 0, stream>>>(x, qkv_w, proj_w, X16, W16, PW16);
  k_gemm_qkv<<<dim3(36, 65), 256, 0, stream>>>(X16, W16, qkv_b, Q16, K16, VT16);
  k_qt<<<(B_ * NH * SEQ * NBUK + 255) / 256, 256, 0, stream>>>(Q16, rpe_r, rpe_c,
                                                               QTR, QTC, nb);
  k_flash<<<B_ * NH * 65, 32, 0, stream>>>(Q16, K16, VT16, QTR, QTC, brow, bcol,
                                           AO16);
  k_gemm_proj<<<dim3(12, 65), 256, 0, stream>>>(AO16, PW16, proj_b, (float*)d_out);
}
